// STAMP_6098853561019
// MI455X (gfx1250) — compile-verified, matrix operand order checked by reading, not run on hardware
//
#include <hip/hip_runtime.h>

#define T_DIM 200
#define B_DIM 64
#define E_DIM 64
#define H_DIM 64
#define ROWS  (T_DIM * B_DIM)   // 12800
#define IT    16                // i-tile
#define JC    32                // j-chunk (WMMA K)
#define NW    2                 // waves per block in kernel C
#define PAD   68                // LDS row stride (floats), conflict-free

#define USE_ASYNC_STAGE 1       // gfx1250 GLOBAL_LOAD_ASYNC_TO_LDS staging

#define NEG_L2E (-1.44269504088896340736f)

typedef __attribute__((ext_vector_type(16))) __bf16          v16bf;
typedef __attribute__((ext_vector_type(8)))  float           v8f;

__device__ __forceinline__ float leaky(float x) {
    return (x >= 0.0f) ? x : 0.3f * x;
}

__device__ __forceinline__ float fast_sigmoid(float x) {
    float e = __builtin_amdgcn_exp2f(x * NEG_L2E);
    return __builtin_amdgcn_rcpf(1.0f + e);
}

// ---------------------------------------------------------------------------
// Kernel A: running mean over time.  4096 threads, each scans T.
// ---------------------------------------------------------------------------
__global__ void ms_cumsum_kernel(const float* __restrict__ in,
                                 float* __restrict__ ms) {
    int id = blockIdx.x * blockDim.x + threadIdx.x;   // (b,e) flat, 0..4095
    float sum = 0.0f;
    for (int t = 0; t < T_DIM; ++t) {
        sum += in[t * (B_DIM * E_DIM) + id];
        ms[t * (B_DIM * E_DIM) + id] = sum / (float)(t + 1);
    }
}

// ---------------------------------------------------------------------------
// Kernel B: x1 = in@w1 + b1 ; q = in@w2 + m_s@w3.
// Both are consumed ONLY inside sigmoid(x1[j]+q[i]) -> store pre-scaled by
// -log2(e) so kernel C's inner loop is a bare v_exp_f32.
// ---------------------------------------------------------------------------
__global__ __launch_bounds__(256) void x1q_kernel(
        const float* __restrict__ in, const float* __restrict__ ms,
        const float* __restrict__ w1, const float* __restrict__ b1,
        const float* __restrict__ w2, const float* __restrict__ w3,
        float* __restrict__ x1, float* __restrict__ q) {
    __shared__ float sIn[4][64];
    __shared__ float sMs[4][64];
    const int h   = threadIdx.x & 63;
    const int r   = threadIdx.x >> 6;
    const int row = blockIdx.x * 4 + r;
    sIn[r][h] = in[row * 64 + h];
    sMs[r][h] = ms[row * 64 + h];
    __syncthreads();
    float a  = b1[h];
    float qq = 0.0f;
#pragma unroll 8
    for (int k = 0; k < 64; ++k) {
        const float iv = sIn[r][k];
        const float mv = sMs[r][k];
        a  += iv * w1[k * 64 + h];
        qq += iv * w2[k * 64 + h] + mv * w3[k * 64 + h];
    }
    x1[row * 64 + h] = a * NEG_L2E;    // pre-scaled logits
    q[row * 64 + h]  = qq * NEG_L2E;
}

// ---------------------------------------------------------------------------
// Kernel C: fused masked pairwise sigmoid-score + WMMA contraction.
//   block = (i-tile of 16, batch b), 2 waves; wave w handles j-chunks w, w+2...
//   scores -> bf16 A fragment, inputs tile -> bf16 B fragment,
//   m_a tile (16 x 64) accumulated in 4 v_wmma_f32_16x16x32_bf16 C fragments.
// ---------------------------------------------------------------------------
__global__ __launch_bounds__(64, 1) void score_ma_kernel(
        const float* __restrict__ in, const float* __restrict__ x1g,
        const float* __restrict__ qg, const float* __restrict__ w0,
        float* __restrict__ mag) {
    __shared__ __align__(16) float sQ[IT * PAD];
    __shared__ float sW0[64];
    __shared__ __align__(16) float sX1[NW][JC * PAD];
    __shared__ __align__(16) float sXin[NW][JC * PAD];
    __shared__ float sPart[NW][IT * 64];

    const int tid  = threadIdx.x;
    const int lane = tid & 31;
    const int w    = tid >> 5;
    const int i0   = blockIdx.x * IT;
    const int b    = blockIdx.y;

    // Stage q tile (clamped rows) + w0
    for (int idx = tid; idx < IT * 64; idx += 64) {
        const int row = idx >> 6, h = idx & 63;
        int gi = i0 + row; if (gi > T_DIM - 1) gi = T_DIM - 1;
        sQ[row * PAD + h] = qg[(gi * B_DIM + b) * H_DIM + h];
    }
    if (tid < 64) sW0[tid] = w0[tid];
    __syncthreads();

    const int m     = lane & 15;   // i-row within tile (A-matrix M)
    const int hw    = lane >> 4;   // wave half
    const int n     = m;           // B/C column index for this lane
    const int i_row = i0 + m;

    int max_j = i0 + IT - 1; if (max_j > T_DIM - 1) max_j = T_DIM - 1;
    const int n_chunks = max_j / JC + 1;

    v8f cacc[4];
#pragma unroll
    for (int ec = 0; ec < 4; ++ec)
#pragma unroll
        for (int r8 = 0; r8 < 8; ++r8) cacc[ec][r8] = 0.0f;

    for (int c = w; c < n_chunks; c += NW) {
        const int j0 = c * JC;

        // ---- Per-wave staging: 32 rows x 64 floats of x1' and inputs ----
#if USE_ASYNC_STAGE
        for (int f4 = lane; f4 < JC * 16; f4 += 32) {
            const int row = f4 >> 4, c4 = f4 & 15;
            int gj = j0 + row; if (gj > T_DIM - 1) gj = T_DIM - 1;
            const unsigned goff =
                (unsigned)(((gj * B_DIM + b) * H_DIM + c4 * 4) * sizeof(float));
            const unsigned l1 =
                (unsigned)(uintptr_t)&sX1[w][row * PAD + c4 * 4];
            const unsigned l2 =
                (unsigned)(uintptr_t)&sXin[w][row * PAD + c4 * 4];
            asm volatile("global_load_async_to_lds_b128 %0, %1, %2"
                         :: "v"(l1), "v"(goff), "s"(x1g) : "memory");
            asm volatile("global_load_async_to_lds_b128 %0, %1, %2"
                         :: "v"(l2), "v"(goff), "s"(in) : "memory");
        }
        asm volatile("s_wait_asynccnt 0x0" ::: "memory");
#else
        for (int f4 = lane; f4 < JC * 16; f4 += 32) {
            const int row = f4 >> 4, c4 = f4 & 15;
            int gj = j0 + row; if (gj > T_DIM - 1) gj = T_DIM - 1;
            const float4 vx = *reinterpret_cast<const float4*>(
                &x1g[(gj * B_DIM + b) * H_DIM + c4 * 4]);
            const float4 vi = *reinterpret_cast<const float4*>(
                &in[(gj * B_DIM + b) * E_DIM + c4 * 4]);
            *reinterpret_cast<float4*>(&sX1[w][row * PAD + c4 * 4])  = vx;
            *reinterpret_cast<float4*>(&sXin[w][row * PAD + c4 * 4]) = vi;
        }
#endif
        __builtin_amdgcn_wave_barrier();   // same-wave DS ops are in-order

        // 16 scores per lane, laid out as the bf16 16x32 A fragment:
        // half 0 lanes: K in {0..7, 16..23}; half 1 lanes: K in {8..15, 24..31}
        float acc[16];
#pragma unroll
        for (int k = 0; k < 16; ++k) acc[k] = 0.0f;

        for (int h = 0; h < H_DIM; ++h) {
            const float qv  = sQ[m * PAD + h];   // per-lane bank-distinct
            const float w0h = sW0[h];            // broadcast
#pragma unroll
            for (int k = 0; k < 16; ++k) {
                const int K = (k < 8) ? (hw * 8 + k) : (16 + hw * 8 + (k - 8));
                const float xv = sX1[w][K * PAD + h];   // half-broadcast
                // logits are pre-scaled by -log2(e): exp2 -> 1+e -> rcp
                const float e = __builtin_amdgcn_exp2f(xv + qv);
                acc[k] += w0h * __builtin_amdgcn_rcpf(1.0f + e);
            }
        }

        // Mask (j <= i, in-range) in the data, keep EXEC all-1s; -> bf16
        v16bf afrag;
#pragma unroll
        for (int k = 0; k < 16; ++k) {
            const int K = (k < 8) ? (hw * 8 + k) : (16 + hw * 8 + (k - 8));
            const int j = j0 + K;
            const bool valid = (j <= i_row) && (i_row < T_DIM);
            afrag[k] = (__bf16)(valid ? acc[k] : 0.0f);
        }

#pragma unroll
        for (int ec = 0; ec < 4; ++ec) {
            v16bf bfrag;
#pragma unroll
            for (int e8 = 0; e8 < 16; ++e8) {
                const int K = (e8 < 8) ? (hw * 8 + e8) : (16 + hw * 8 + (e8 - 8));
                bfrag[e8] = (__bf16)sXin[w][K * PAD + ec * 16 + n];
            }
            cacc[ec] = __builtin_amdgcn_wmma_f32_16x16x32_bf16(
                false, afrag, false, bfrag, (short)0, cacc[ec], false, false);
        }
    }

    // C/D layout: lane hw*16+n, VGPR r -> M = hw*8+r, N = n
#pragma unroll
    for (int ec = 0; ec < 4; ++ec)
#pragma unroll
        for (int r8 = 0; r8 < 8; ++r8)
            sPart[w][(hw * 8 + r8) * 64 + ec * 16 + n] = cacc[ec][r8];
    __syncthreads();

    for (int idx = tid; idx < IT * 64; idx += 64) {
        const int il = idx >> 6, e = idx & 63;
        const int i = i0 + il;
        if (i < T_DIM)
            mag[(i * B_DIM + b) * E_DIM + e] = sPart[0][idx] + sPart[1][idx];
    }
}

// ---------------------------------------------------------------------------
// Kernel D: two 2-layer LeakyReLU towers per head (pc, pv), weights in LDS.
// ---------------------------------------------------------------------------
struct PredW {
    const float* a0k; const float* a0b; const float* b0k; const float* b0b;
    const float* a1k; const float* a1b; const float* b1k; const float* b1b;
};

#define OFF_A0K 0
#define OFF_B0K 4096
#define OFF_A1K 8192
#define OFF_B1K 10240
#define OFF_A0B 12288
#define OFF_B0B 12352
#define OFF_A1B 12416
#define OFF_B1B 12448
#define SW_TOT  12480

__global__ __launch_bounds__(256) void pred_kernel(
        const float* __restrict__ in, const float* __restrict__ mag,
        PredW wc, PredW wv, float* __restrict__ out) {
    __shared__ float sw[SW_TOT];
    const int tid = threadIdx.x;
    const int row = blockIdx.x * 256 + tid;

    float res[2];
    for (int p = 0; p < 2; ++p) {
        const PredW W = p ? wv : wc;
        __syncthreads();
        for (int idx = tid; idx < SW_TOT; idx += 256) {
            float v;
            if      (idx < OFF_B0K) v = W.a0k[idx - OFF_A0K];
            else if (idx < OFF_A1K) v = W.b0k[idx - OFF_B0K];
            else if (idx < OFF_B1K) v = W.a1k[idx - OFF_A1K];
            else if (idx < OFF_A0B) v = W.b1k[idx - OFF_B1K];
            else if (idx < OFF_B0B) v = W.a0b[idx - OFF_A0B];
            else if (idx < OFF_A1B) v = W.b0b[idx - OFF_B0B];
            else if (idx < OFF_B1B) v = W.a1b[idx - OFF_A1B];
            else                    v = W.b1b[idx - OFF_B1B];
            sw[idx] = v;
        }
        __syncthreads();

        float x[64], h1[64], ha2[32], hb2[32];

        // tower a on m_a
        for (int k = 0; k < 64; ++k) x[k] = mag[row * 64 + k];
        for (int o = 0; o < 64; ++o) {
            float acc = sw[OFF_A0B + o];
#pragma unroll 8
            for (int k = 0; k < 64; ++k) acc += x[k] * sw[OFF_A0K + k * 64 + o];
            h1[o] = leaky(acc);
        }
        for (int o = 0; o < 32; ++o) {
            float acc = sw[OFF_A1B + o];
#pragma unroll 8
            for (int k = 0; k < 64; ++k) acc += h1[k] * sw[OFF_A1K + k * 32 + o];
            ha2[o] = leaky(acc);
        }
        // tower b on inputs
        for (int k = 0; k < 64; ++k) x[k] = in[row * 64 + k];
        for (int o = 0; o < 64; ++o) {
            float acc = sw[OFF_B0B + o];
#pragma unroll 8
            for (int k = 0; k < 64; ++k) acc += x[k] * sw[OFF_B0K + k * 64 + o];
            h1[o] = leaky(acc);
        }
        for (int o = 0; o < 32; ++o) {
            float acc = sw[OFF_B1B + o];
#pragma unroll 8
            for (int k = 0; k < 64; ++k) acc += h1[k] * sw[OFF_B1K + k * 32 + o];
            hb2[o] = leaky(acc);
        }
        float d = 0.0f;
#pragma unroll
        for (int o = 0; o < 32; ++o) d += ha2[o] * hb2[o];
        res[p] = fast_sigmoid(d);
    }
    out[row]        = res[0];            // pc
    out[ROWS + row] = res[1] * res[0];   // pv = pv * pc
}

// ---------------------------------------------------------------------------
extern "C" void kernel_launch(void* const* d_in, const int* in_sizes, int n_in,
                              void* d_out, int out_size, void* d_ws, size_t ws_size,
                              hipStream_t stream) {
    const float* in = (const float*)d_in[0];
    const float* w1 = (const float*)d_in[1];
    const float* b1 = (const float*)d_in[2];
    const float* w2 = (const float*)d_in[3];
    const float* w3 = (const float*)d_in[4];
    const float* w0 = (const float*)d_in[5];
    PredW wc{(const float*)d_in[6],  (const float*)d_in[7],
             (const float*)d_in[8],  (const float*)d_in[9],
             (const float*)d_in[10], (const float*)d_in[11],
             (const float*)d_in[12], (const float*)d_in[13]};
    PredW wv{(const float*)d_in[14], (const float*)d_in[15],
             (const float*)d_in[16], (const float*)d_in[17],
             (const float*)d_in[18], (const float*)d_in[19],
             (const float*)d_in[20], (const float*)d_in[21]};

    float* ws  = (float*)d_ws;
    float* ms  = ws;                    // [T,B,E]
    float* x1  = ws + 1 * ROWS * 64;    // [T,B,H] pre-scaled by -log2(e)
    float* q   = ws + 2 * ROWS * 64;    // [T,B,H] pre-scaled by -log2(e)
    float* ma  = ws + 3 * ROWS * 64;    // [T,B,E]
    float* out = (float*)d_out;

    ms_cumsum_kernel<<<16, 256, 0, stream>>>(in, ms);
    x1q_kernel<<<ROWS / 4, 256, 0, stream>>>(in, ms, w1, b1, w2, w3, x1, q);
    score_ma_kernel<<<dim3((T_DIM + IT - 1) / IT, B_DIM), NW * 32, 0, stream>>>(
        in, x1, q, w0, ma);
    pred_kernel<<<ROWS / 256, 256, 0, stream>>>(in, ma, wc, wv, out);
}
